// DynamicConv2d_33165737460341
// MI455X (gfx1250) — matrix operand order repaired and checked by reading, not compiled, not run on hardware
//
#include <hip/hip_runtime.h>

// ---------------------------------------------------------------------------
// DynamicConv2d for MI455X (gfx1250, wave32, WMMA).
// Dominant cost: per-sample 3x3 conv == per-batch GEMM M=64, N=65536, K=576.
// Memory-bound at 23.3 TB/s (~2.7 GB moved) provided math runs on
// v_wmma_f32_16x16x32_f16; inputs converted to f16, accumulate in f32.
// ---------------------------------------------------------------------------

typedef _Float16 half_t;
typedef __attribute__((ext_vector_type(16))) _Float16 v16h;
typedef __attribute__((ext_vector_type(8)))  _Float16 v8h;
typedef __attribute__((ext_vector_type(8)))  float    v8f;

#define B_    32
#define CIN   64
#define COUT  64
#define H_    256
#define W_    256
#define KEXP  4
#define KTOT  576      // 9 * CIN, GEMM K
#define WTILE 128      // output columns per block
#define XCOLS 130      // WTILE + 2 halo columns

// ---------------- kernel 1: global average pool g[b,c] ----------------
__global__ void pool_kernel(const float* __restrict__ x, float* __restrict__ g) {
  __shared__ float red[256];
  const int bc = blockIdx.x;                       // b*CIN + c
  const float* p = x + (size_t)bc * (H_ * W_);
  float s = 0.f;
  for (int i = threadIdx.x; i < H_ * W_; i += 256) s += p[i];
  red[threadIdx.x] = s;
  __syncthreads();
  for (int off = 128; off > 0; off >>= 1) {
    if (threadIdx.x < off) red[threadIdx.x] += red[threadIdx.x + off];
    __syncthreads();
  }
  if (threadIdx.x == 0) g[bc] = red[0] * (1.0f / (H_ * W_));
}

// ---------------- kernel 2: MLP + softmax -> a[b,k] ----------------
__global__ void attn_kernel(const float* __restrict__ g,
                            const float* __restrict__ w1, const float* __restrict__ b1,
                            const float* __restrict__ w2, const float* __restrict__ b2,
                            float* __restrict__ a) {
  const int b = blockIdx.x;        // one wave32 per sample
  __shared__ float hsh[32];
  __shared__ float lg[KEXP];
  const int j = threadIdx.x;       // 0..31
  float hv = b1[j];
  for (int c = 0; c < CIN; ++c) hv += g[b * CIN + c] * w1[j * CIN + c];
  hsh[j] = fmaxf(hv, 0.f);
  __syncthreads();
  if (j < KEXP) {
    float lv = b2[j];
    for (int t = 0; t < 32; ++t) lv += hsh[t] * w2[j * 32 + t];
    lg[j] = lv;                     // TAU == 1.0
  }
  __syncthreads();
  if (j == 0) {
    float m = lg[0];
    for (int k = 1; k < KEXP; ++k) m = fmaxf(m, lg[k]);
    float e[KEXP], s = 0.f;
    for (int k = 0; k < KEXP; ++k) { e[k] = __expf(lg[k] - m); s += e[k]; }
    for (int k = 0; k < KEXP; ++k) a[b * KEXP + k] = e[k] / s;
  }
}

// ---- kernel 3: mix expert weights -> f16 [b][co][r*64+ci], mixed bias ----
__global__ void mixw_kernel(const float* __restrict__ weight,
                            const float* __restrict__ bias,
                            const float* __restrict__ a,
                            half_t* __restrict__ wmixh, float* __restrict__ bmix) {
  const int idx = blockIdx.x * 256 + threadIdx.x;  // < B*COUT*KTOT
  const int ci = idx % CIN;
  int t = idx / CIN;
  const int r  = t % 9;            // r = kh*3 + kw
  t /= 9;
  const int co = t % COUT;
  const int b  = t / COUT;
  const float* ab = a + b * KEXP;
  float s = 0.f;
#pragma unroll
  for (int k = 0; k < KEXP; ++k)
    s += ab[k] * weight[((k * COUT + co) * CIN + ci) * 9 + r];
  wmixh[idx] = (half_t)s;
  if (idx < B_ * COUT) {
    const int bb = idx / COUT, oo = idx % COUT;
    float bs = 0.f;
#pragma unroll
    for (int k = 0; k < KEXP; ++k) bs += a[bb * KEXP + k] * bias[k * COUT + oo];
    bmix[idx] = bs;
  }
}

// ---------------- kernel 4: implicit-GEMM conv via WMMA ----------------
// Block: 256 thr (8 waves). Tile: batch b, row h, cols [w0, w0+128), all 64 Cout.
// LDS: weights [64][576] f16 (72 KB) + x halo [3][130][64] f16 (48.75 KB).
// K ordering: k = (kh*3+kw)*64 + ci  -> each 32-chunk is fixed (kh,kw),
// 32 consecutive channels => contiguous 16B LDS fragment loads.
__global__ __launch_bounds__(256)
void conv_kernel(const float* __restrict__ x, const half_t* __restrict__ wmixh,
                 const float* __restrict__ bmix, float* __restrict__ y) {
  extern __shared__ char smem[];
  half_t* wt = (half_t*)smem;                               // [COUT][KTOT]
  half_t* xt = (half_t*)(smem + COUT * KTOT * sizeof(half_t)); // [3][XCOLS][CIN]

  const int w0  = blockIdx.x * WTILE;
  const int h   = blockIdx.y;
  const int b   = blockIdx.z;
  const int tid = threadIdx.x;

  // load per-batch weight matrix (16B vectors)
  {
    const uint4* src = (const uint4*)(wmixh + (size_t)b * COUT * KTOT);
    uint4* dst = (uint4*)wt;
    for (int i = tid; i < COUT * KTOT / 8; i += 256) dst[i] = src[i];
  }
  // load x halo tile with zero padding, f32 -> f16, channel-innermost
  {
    const int total = 3 * CIN * XCOLS;
    for (int e = tid; e < total; e += 256) {
      const int col = e % XCOLS;
      int t = e / XCOLS;
      const int ci  = t % CIN;
      const int row = t / CIN;
      const int hh = h + row - 1;
      const int ww = w0 + col - 1;
      float v = 0.f;
      if ((unsigned)hh < H_ && (unsigned)ww < W_)
        v = x[(((size_t)b * CIN + ci) * H_ + hh) * W_ + ww];
      xt[(row * XCOLS + col) * CIN + ci] = (half_t)v;
    }
  }
  __syncthreads();

  const int wave    = tid >> 5;
  const int lane    = tid & 31;
  const int laneLow = lane & 15;
  const int laneHi  = lane >> 4;          // 0/1: second half of wave
  const int mBase   = (wave & 3) * 16;    // Cout tile
  const int nGrp    = (wave >> 2) * 64;   // 4 N-tiles of 16 per wave

  v8f acc[4] = {};
  const int coA     = mBase + laneLow;    // A: M striped over lanes 0-15/16-31
  const int khalfA  = laneHi * 8;         // A: lanes>=16 shifted by K=8
  const int ciHalfB = laneHi * 8;         // B: lanes>=16 shifted by K=8

#pragma unroll
  for (int kc = 0; kc < 18; ++kc) {       // 576 / 32 K-chunks
    const int r      = kc >> 1;           // two chunks per (kh,kw)
    const int kh     = r / 3, kw = r % 3;
    const int ciBase = (kc & 1) * 32;

    // A fragment: wt[coA][kc*32 + klocal], layout per ISA 16-bit A 16x32
    const half_t* ap = wt + coA * KTOT + kc * 32;
    v8h alo = *(const v8h*)(ap + khalfA);        // elems 0..7  (K 0-7 | 8-15)
    v8h ahi = *(const v8h*)(ap + 16 + khalfA);   // elems 8..15 (K 16-23 | 24-31)
    v16h A;
#pragma unroll
    for (int i = 0; i < 8; ++i) { A[i] = alo[i]; A[i + 8] = ahi[i]; }

#pragma unroll
    for (int j = 0; j < 4; ++j) {
      // B fragment: x at fixed (kh,kw), N = spatial column, K = channel
      const int col = nGrp + j * 16 + laneLow + kw;       // halo-local column
      const half_t* bp = xt + (kh * XCOLS + col) * CIN + ciBase + ciHalfB;
      v8h blo = *(const v8h*)bp;
      v8h bhi = *(const v8h*)(bp + 16);
      v16h Bf;
#pragma unroll
      for (int i = 0; i < 8; ++i) { Bf[i] = blo[i]; Bf[i + 8] = bhi[i]; }

      acc[j] = __builtin_amdgcn_wmma_f32_16x16x32_f16(
          /*neg_a=*/false, A, /*neg_b=*/false, Bf,
          /*c_mod=*/(short)0, acc[j], /*reuse_a=*/false, /*reuse_b=*/false);
    }
  }

  // bias per output channel handled by this lane (C/D: M = v + 8*laneHi)
  float bv[8];
#pragma unroll
  for (int v = 0; v < 8; ++v)
    bv[v] = bmix[b * COUT + mBase + v + laneHi * 8];

#pragma unroll
  for (int j = 0; j < 4; ++j) {
    const int wcol = w0 + nGrp + j * 16 + laneLow;        // N = laneLow
#pragma unroll
    for (int v = 0; v < 8; ++v) {
      const int co = mBase + v + laneHi * 8;
      y[(((size_t)b * COUT + co) * H_ + h) * W_ + wcol] = acc[j][v] + bv[v];
    }
  }
}

// ---------------------------------------------------------------------------
extern "C" void kernel_launch(void* const* d_in, const int* in_sizes, int n_in,
                              void* d_out, int out_size, void* d_ws, size_t ws_size,
                              hipStream_t stream) {
  const float* x      = (const float*)d_in[0];
  const float* weight = (const float*)d_in[1];
  const float* bias   = (const float*)d_in[2];
  const float* w1     = (const float*)d_in[3];
  const float* b1     = (const float*)d_in[4];
  const float* w2     = (const float*)d_in[5];
  const float* b2     = (const float*)d_in[6];
  float* y = (float*)d_out;

  char* ws = (char*)d_ws;
  float*  g     = (float*)ws;                 // B*CIN  = 2048 f32
  float*  a     = (float*)(ws + 8192);        // B*KEXP = 128 f32
  float*  bmix  = (float*)(ws + 8704);        // B*COUT = 2048 f32
  half_t* wmixh = (half_t*)(ws + 16896);      // B*COUT*KTOT = 1179648 f16

  pool_kernel<<<B_ * CIN, 256, 0, stream>>>(x, g);
  attn_kernel<<<B_, 32, 0, stream>>>(g, w1, b1, w2, b2, a);
  mixw_kernel<<<(B_ * COUT * KTOT) / 256, 256, 0, stream>>>(weight, bias, a, wmixh, bmix);

  dim3 grid(W_ / WTILE, H_, B_);              // 2 x 256 x 32 blocks
  const size_t smem = (size_t)COUT * KTOT * 2 + (size_t)3 * XCOLS * CIN * 2; // 123648 B
  conv_kernel<<<grid, 256, smem, stream>>>(x, wmixh, bmix, y);
}
